// AlgebraicTransformerLM_41068477284953
// MI455X (gfx1250) — matrix operand order repaired and checked by reading, not compiled, not run on hardware
//
#include <hip/hip_runtime.h>
#include <hip/hip_bf16.h>

// ---------------- model dims (fixed by reference) ----------------
constexpr int Lc = 4, Bc = 2, Tc = 1024, Dc = 1024, Hc = 16, DHc = 64,
              Fc = 4096, Vc = 32000;
constexpr float EPSc = 1e-6f;

typedef __bf16 bf16_t;
typedef __attribute__((ext_vector_type(16))) __bf16 v16bf;
typedef __attribute__((ext_vector_type(8)))  float  v8f;

union FragU { uint4 u[2]; v16bf v; };

// ---------------- small helpers ----------------
__device__ __forceinline__ float blockReduceSum(float v, float* sh) {
  int t = threadIdx.x;
  sh[t] = v;
  __syncthreads();
  for (int s = 128; s > 0; s >>= 1) {
    if (t < s) sh[t] += sh[t + s];
    __syncthreads();
  }
  float r = sh[0];
  __syncthreads();
  return r;
}

// ---------------- weight convert + transpose: f32 [K,N] -> bf16 [N,K] -------
__global__ void k_cvt_transpose(const float* __restrict__ src,
                                bf16_t* __restrict__ dst, int K, int N) {
  long i = (long)blockIdx.x * blockDim.x + threadIdx.x;
  long total = (long)K * N;
  if (i >= total) return;
  int k = (int)(i / N);
  int n = (int)(i % N);
  dst[(long)n * K + k] = (bf16_t)src[i];
}

// ---------------- embedding: x[b,t,:] = tok_emb[idx[b,t],:] + pos_emb[t,:] --
__global__ void k_embed(const int* __restrict__ idx,
                        const float* __restrict__ tok,
                        const float* __restrict__ pos,
                        float* __restrict__ x) {
  int row = blockIdx.x;                 // b*T + t
  int t = row % Tc;
  long tokid = idx[row];
  const float* te = tok + tokid * (long)Dc;
  const float* pe = pos + (long)t * Dc;
  float* xr = x + (long)row * Dc;
  for (int i = threadIdx.x; i < Dc; i += blockDim.x) xr[i] = te[i] + pe[i];
}

// ---------------- L1 anorm -> bf16: h = g * x / (mean|x| + eps) -------------
__global__ void k_anorm(const float* __restrict__ x,
                        const float* __restrict__ gain,
                        bf16_t* __restrict__ h) {
  __shared__ float sh[256];
  int row = blockIdx.x;                 // b*T + t
  const float* xr = x + (long)row * Dc;
  float vals[4];
  float s = 0.f;
#pragma unroll
  for (int j = 0; j < 4; j++) {
    vals[j] = xr[threadIdx.x + j * 256];
    s += fabsf(vals[j]);
  }
  float tot = blockReduceSum(s, sh);
  float inv = 1.0f / (tot * (1.0f / Dc) + EPSc);
  bf16_t* hr = h + (long)row * Dc;
#pragma unroll
  for (int j = 0; j < 4; j++) {
    int i = threadIdx.x + j * 256;
    hr[i] = (bf16_t)(gain[i] * vals[j] * inv);
  }
}

// ---------------- score row post-process + rational softmax -> bf16 ---------
__global__ void k_score_post(const float* __restrict__ scores,
                             bf16_t* __restrict__ attn,
                             const float* __restrict__ score_gain_l) {
  __shared__ float sh[256];
  long row = blockIdx.x;                // over B*H*T
  int tq = (int)(row % Tc);
  const float* sr = scores + row * Tc;
  float v[4];
  float s = 0.f;
#pragma unroll
  for (int j = 0; j < 4; j++) {
    v[j] = sr[threadIdx.x + j * 256];
    s += v[j];
  }
  float mean = blockReduceSum(s, sh) * (1.0f / Tc);
  s = 0.f;
#pragma unroll
  for (int j = 0; j < 4; j++) {
    v[j] -= mean;
    s += fabsf(v[j]);
  }
  float mad = blockReduceSum(s, sh) * (1.0f / Tc) + EPSc;
  float g = score_gain_l[0] / mad;
  float p[4];
  float psum = 0.f;
#pragma unroll
  for (int j = 0; j < 4; j++) {
    int col = threadIdx.x + j * 256;
    float sc = v[j] * g;
    if (col > tq) sc = -1000.0f;
    float sq = sc / (fabsf(sc) + 1.0f);
    float u = (sq + 1.0f) * 0.5f;
    u = u * u;
    u = u * u;                          // ((s+1)/2)^4
    p[j] = u;
    psum += u;
  }
  float denom = 1.0f / (blockReduceSum(psum, sh) + EPSc);
  bf16_t* ar = attn + row * Tc;
#pragma unroll
  for (int j = 0; j < 4; j++)
    ar[threadIdx.x + j * 256] = (bf16_t)(p[j] * denom);
}

// ---------------- generic bf16 WMMA GEMM (software-pipelined) ---------------
// C[M,N] = A[M,K] (row-major) x Bt[N,K] (row-major, i.e. B transposed)
// 8 waves / block. Each wave: TWO 16-row tiles x four 16-col tiles (32x64).
// wavesN in {1,4}: wave grid inside block is (8/wavesN) x wavesN.
// K must be a multiple of 64 (true for all uses: 64, 1024, 4096).
// Register double-buffered, branchless ping-pong: loads for step i+1 are
// issued before the 8 WMMAs of step i; global_prefetch hints run 2 steps ahead.
enum { M_F32 = 0, M_QK = 1, M_VT = 2, M_O = 3, M_RESID = 4, M_RELU = 5 };

__device__ __forceinline__ void load_frags(const bf16_t* ar0,
                                           const bf16_t* ar1,
                                           const bf16_t* const br[4], int kk,
                                           FragU a[2], FragU b[4]) {
  a[0].u[0] = *(const uint4*)(ar0 + kk);
  a[0].u[1] = *(const uint4*)(ar0 + kk + 16);
  a[1].u[0] = *(const uint4*)(ar1 + kk);
  a[1].u[1] = *(const uint4*)(ar1 + kk + 16);
#pragma unroll
  for (int j = 0; j < 4; j++) {
    b[j].u[0] = *(const uint4*)(br[j] + kk);
    b[j].u[1] = *(const uint4*)(br[j] + kk + 16);
  }
}

__device__ __forceinline__ void do_wmma(const FragU a[2], const FragU b[4],
                                        v8f acc[2][4]) {
#pragma unroll
  for (int mi = 0; mi < 2; mi++)
#pragma unroll
    for (int j = 0; j < 4; j++)
      acc[mi][j] = __builtin_amdgcn_wmma_f32_16x16x32_bf16(
          false, a[mi].v, false, b[j].v, (short)0, acc[mi][j], false, false);
}

__global__ void __launch_bounds__(256, 1)
k_gemm(const bf16_t* __restrict__ A, long sA, int lda,
       const bf16_t* __restrict__ Bt, long sB, int ldb,
       const float* __restrict__ bias,
       float* __restrict__ Cf, bf16_t* __restrict__ Cb,
       long sC, int ldc, int K, int mode, int wavesN) {
  int z = blockIdx.z;
  A += (long)z * sA;
  Bt += (long)z * sB;

  int lane = threadIdx.x & 31;
  int lane_lo = lane & 15;
  int lane_hi = lane >> 4;
  int w = threadIdx.x >> 5;
  int wm = w / wavesN;
  int wn = w - wm * wavesN;
  int wavesM = 8 / wavesN;
  int m0 = (blockIdx.y * wavesM + wm) * 32;   // 2 m-tiles per wave
  int n0 = (blockIdx.x * wavesN + wn) * 64;   // 4 n-tiles per wave

  v8f acc[2][4] = {{v8f{}, v8f{}, v8f{}, v8f{}},
                   {v8f{}, v8f{}, v8f{}, v8f{}}};

  // Per-lane fragment base pointers (A and B: two 8-elem K-chunks)
  const bf16_t* ar0 = A + (long)(m0 + lane_lo) * lda + lane_hi * 8;
  const bf16_t* ar1 = A + (long)(m0 + 16 + lane_lo) * lda + lane_hi * 8;
  const bf16_t* br[4];
#pragma unroll
  for (int j = 0; j < 4; j++)
    br[j] = Bt + (long)(n0 + j * 16 + lane_lo) * ldb + lane_hi * 8;

  FragU a0[2], b0[4], a1[2], b1[4];
  load_frags(ar0, ar1, br, 0, a0, b0);
  for (int kk = 0; kk < K; kk += 64) {
    load_frags(ar0, ar1, br, kk + 32, a1, b1);
    // L2-directed prefetch two K-steps ahead (speculative; clamped in-range)
    int kp = kk + 128;
    kp = (kp < K) ? kp : 0;
    __builtin_prefetch(ar0 + kp, 0, 3);
    __builtin_prefetch(ar1 + kp, 0, 3);
#pragma unroll
    for (int j = 0; j < 4; j++) __builtin_prefetch(br[j] + kp, 0, 3);
    do_wmma(a0, b0, acc);
    int kn = kk + 64;
    kn = (kn < K) ? kn : 0;              // clamp: last prefetch is redundant
    load_frags(ar0, ar1, br, kn, a0, b0);
    do_wmma(a1, b1, acc);
  }

  // ---- epilogue: C/D layout: VGPR r -> row base + r + 8*lane_hi, col lane_lo
#pragma unroll
  for (int mi = 0; mi < 2; mi++) {
#pragma unroll
    for (int j = 0; j < 4; j++) {
#pragma unroll
      for (int r = 0; r < 8; r++) {
        int m = m0 + mi * 16 + r + lane_hi * 8;
        int n = n0 + j * 16 + lane_lo;
        float val = acc[mi][j][r];
        if (mode == M_F32) {
          if (bias) val += bias[n];
          Cf[(long)z * sC + (long)m * ldc + n] = val;
        } else if (mode == M_QK) {
          val += bias[n];
          int b = m >> 10, t = m & (Tc - 1);
          int hh = n >> 6, dh = n & 63;
          Cb[((((long)(b * Hc + hh)) * Tc + t) << 6) + dh] = (bf16_t)val;
        } else if (mode == M_VT) {
          val += bias[n];
          int b = m >> 10, t = m & (Tc - 1);
          int hh = n >> 6, dh = n & 63;
          Cb[((long)(b * Hc + hh) * DHc + dh) * (long)Tc + t] = (bf16_t)val;
        } else if (mode == M_O) {
          int b = z / Hc, hh = z % Hc;
          Cb[((long)(b * Tc + m)) * Dc + hh * DHc + n] = (bf16_t)val;
        } else if (mode == M_RESID) {
          float* p = Cf + (long)m * ldc + n;
          *p = *p + val + bias[n];
        } else {  // M_RELU
          val += bias[n];
          Cb[(long)m * ldc + n] = (bf16_t)fmaxf(val, 0.0f);
        }
      }
    }
  }
}

// ---------------- host orchestration ----------------------------------------
extern "C" void kernel_launch(void* const* d_in, const int* in_sizes, int n_in,
                              void* d_out, int out_size, void* d_ws,
                              size_t ws_size, hipStream_t stream) {
  (void)in_sizes; (void)n_in; (void)out_size; (void)ws_size;

  const int*   idx        = (const int*)d_in[0];
  const float* tok_emb    = (const float*)d_in[1];
  const float* pos_emb    = (const float*)d_in[2];
  const float* Wq         = (const float*)d_in[3];
  const float* bq         = (const float*)d_in[4];
  const float* Wk         = (const float*)d_in[5];
  const float* bk         = (const float*)d_in[6];
  const float* Wv         = (const float*)d_in[7];
  const float* bv         = (const float*)d_in[8];
  const float* Wo         = (const float*)d_in[9];
  const float* bo         = (const float*)d_in[10];
  const float* score_gain = (const float*)d_in[11];
  const float* norm1_gain = (const float*)d_in[12];
  const float* norm2_gain = (const float*)d_in[13];
  const float* W1         = (const float*)d_in[14];
  const float* b1         = (const float*)d_in[15];
  const float* W2         = (const float*)d_in[16];
  const float* b2         = (const float*)d_in[17];
  const float* final_gain = (const float*)d_in[18];
  const float* Wlm        = (const float*)d_in[19];
  const float* blm        = (const float*)d_in[20];

  // ---- carve workspace
  char* ws = (char*)d_ws;
  size_t off = 0;
  auto alloc = [&](size_t bytes) -> char* {
    char* p = ws + off;
    off += (bytes + 255) & ~(size_t)255;
    return p;
  };
  bf16_t* WqT  = (bf16_t*)alloc((size_t)Lc * Dc * Dc * 2);
  bf16_t* WkT  = (bf16_t*)alloc((size_t)Lc * Dc * Dc * 2);
  bf16_t* WvT  = (bf16_t*)alloc((size_t)Lc * Dc * Dc * 2);
  bf16_t* WoT  = (bf16_t*)alloc((size_t)Lc * Dc * Dc * 2);
  bf16_t* W1T  = (bf16_t*)alloc((size_t)Lc * Dc * Fc * 2);
  bf16_t* W2T  = (bf16_t*)alloc((size_t)Lc * Fc * Dc * 2);
  bf16_t* WlmT = (bf16_t*)alloc((size_t)Dc * Vc * 2);
  float*  x    = (float*)alloc((size_t)Bc * Tc * Dc * 4);
  bf16_t* h    = (bf16_t*)alloc((size_t)Bc * Tc * Dc * 2);
  bf16_t* qb   = (bf16_t*)alloc((size_t)Bc * Tc * Dc * 2);   // [B,H,T,DH]
  bf16_t* kb   = (bf16_t*)alloc((size_t)Bc * Tc * Dc * 2);   // [B,H,T,DH]
  bf16_t* vbT  = (bf16_t*)alloc((size_t)Bc * Tc * Dc * 2);   // [B,H,DH,T]
  bf16_t* obuf = (bf16_t*)alloc((size_t)Bc * Tc * Dc * 2);   // [B*T,D]
  bf16_t* ff   = (bf16_t*)alloc((size_t)Bc * Tc * Fc * 2);   // [B*T,F]
  float*  scr  = (float*)alloc((size_t)Bc * Hc * Tc * Tc * 4); // scores f32
  bf16_t* attn = (bf16_t*)alloc((size_t)Bc * Hc * Tc * Tc * 2);

  auto cvt = [&](const float* src, bf16_t* dst, int K, int N) {
    long total = (long)K * N;
    k_cvt_transpose<<<dim3((unsigned)((total + 255) / 256)), 256, 0, stream>>>(
        src, dst, K, N);
  };
  auto gemm = [&](const bf16_t* Aa, long sA, int lda, const bf16_t* Bb, long sB,
                  int ldb, const float* bias, float* Cf, bf16_t* Cb, long sC,
                  int ldc, int M, int N, int K, int mode, int wavesN,
                  int batch) {
    int wavesM = 8 / wavesN;
    dim3 grid((unsigned)(N / (64 * wavesN)), (unsigned)(M / (32 * wavesM)),
              (unsigned)batch);
    k_gemm<<<grid, 256, 0, stream>>>(Aa, sA, lda, Bb, sB, ldb, bias, Cf, Cb,
                                     sC, ldc, K, mode, wavesN);
  };

  // ---- weight prep (bf16, transposed)
  for (int l = 0; l < Lc; l++) {
    cvt(Wq + (long)l * Dc * Dc, WqT + (long)l * Dc * Dc, Dc, Dc);
    cvt(Wk + (long)l * Dc * Dc, WkT + (long)l * Dc * Dc, Dc, Dc);
    cvt(Wv + (long)l * Dc * Dc, WvT + (long)l * Dc * Dc, Dc, Dc);
    cvt(Wo + (long)l * Dc * Dc, WoT + (long)l * Dc * Dc, Dc, Dc);
    cvt(W1 + (long)l * Dc * Fc, W1T + (long)l * Fc * Dc, Dc, Fc);
    cvt(W2 + (long)l * Fc * Dc, W2T + (long)l * Dc * Fc, Fc, Dc);
  }
  cvt(Wlm, WlmT, Dc, Vc);

  // ---- embedding
  k_embed<<<Bc * Tc, 256, 0, stream>>>(idx, tok_emb, pos_emb, x);

  const int MT = Bc * Tc;  // 2048 token rows
  for (int l = 0; l < Lc; l++) {
    // norm1 -> h (bf16)
    k_anorm<<<MT, 256, 0, stream>>>(x, norm1_gain + (long)l * Dc, h);
    // q, k -> [B,H,T,DH]; v -> [B,H,DH,T]
    gemm(h, 0, Dc, WqT + (long)l * Dc * Dc, 0, Dc, bq + (long)l * Dc, nullptr,
         qb, 0, 0, MT, Dc, Dc, M_QK, 4, 1);
    gemm(h, 0, Dc, WkT + (long)l * Dc * Dc, 0, Dc, bk + (long)l * Dc, nullptr,
         kb, 0, 0, MT, Dc, Dc, M_QK, 4, 1);
    gemm(h, 0, Dc, WvT + (long)l * Dc * Dc, 0, Dc, bv + (long)l * Dc, nullptr,
         vbT, 0, 0, MT, Dc, Dc, M_VT, 4, 1);
    // scores[b,h] = q[T,DH] x k[T,DH]^T   (batched over B*H)
    gemm(qb, (long)Tc * DHc, DHc, kb, (long)Tc * DHc, DHc, nullptr, scr,
         nullptr, (long)Tc * Tc, Tc, Tc, Tc, DHc, M_F32, 4, Bc * Hc);
    // rational softmax etc -> attn (bf16)
    k_score_post<<<Bc * Hc * Tc, 256, 0, stream>>>(scr, attn,
                                                   score_gain + l);
    // o[b,h] = attn[T,T] x v^T[DH,T]^T -> written as [B*T, D]
    gemm(attn, (long)Tc * Tc, Tc, vbT, (long)DHc * Tc, Tc, nullptr, nullptr,
         obuf, 0, 0, Tc, DHc, Tc, M_O, 1, Bc * Hc);
    // x += o @ Wo + bo
    gemm(obuf, 0, Dc, WoT + (long)l * Dc * Dc, 0, Dc, bo + (long)l * Dc, x,
         nullptr, 0, Dc, MT, Dc, Dc, M_RESID, 4, 1);
    // FFN
    k_anorm<<<MT, 256, 0, stream>>>(x, norm2_gain + (long)l * Dc, h);
    gemm(h, 0, Dc, W1T + (long)l * Fc * Dc, 0, Dc, b1 + (long)l * Fc, nullptr,
         ff, 0, Fc, MT, Fc, Dc, M_RELU, 4, 1);
    gemm(ff, 0, Fc, W2T + (long)l * Dc * Fc, 0, Fc, b2 + (long)l * Dc, x,
         nullptr, 0, Dc, MT, Dc, Fc, M_RESID, 4, 1);
  }

  // final norm + LM head -> d_out f32 [B*T, V]
  k_anorm<<<MT, 256, 0, stream>>>(x, final_gain, h);
  gemm(h, 0, Dc, WlmT, 0, Dc, blm, (float*)d_out, nullptr, 0, Vc, MT, Vc, Dc,
       M_F32, 4, 1);
}
